// A2aSparseStackedMlp_72310069396105
// MI455X (gfx1250) — matrix operand order repaired and checked by reading, not compiled, not run on hardware
//
#include <hip/hip_runtime.h>
#include <hip/hip_bf16.h>

typedef __attribute__((ext_vector_type(16))) __bf16 v16bf;
typedef __attribute__((ext_vector_type(8)))  float  v8f;

#define ALPHA 1.702f
#define LIMIT 7.0f

constexpr int T = 2048;   // B*S tokens
constexpr int H = 1024;   // hidden
constexpr int E = 8;      // experts
constexpr int I = 2048;   // intermediate
constexpr int TILE_M = 32;   // tokens per block (two 16-row WMMA halves)
constexpr int NTHREADS = 512; // 16 waves

// ---------------------------------------------------------------------------
// A-fragment (16x32 bf16, MxK) from LDS, per ISA 7.12.2 16-bit A layout:
// lane L: M = L&15 ; lanes>=16 get K+8 ; VGPR v holds K = (v<4 ? 2v : 16+2(v-4))
// m_off selects the M=0..15 (0) or M=16..31 (16) half of the 32-row tile.
// ---------------------------------------------------------------------------
__device__ inline v16bf load_a_lds(const __bf16* __restrict__ base, int stride,
                                   int lane, int k0, int m_off) {
    const __bf16* row = base + ((lane & 15) + m_off) * stride + k0 + ((lane >> 4) << 3);
    v16bf a;
#pragma unroll
    for (int v = 0; v < 8; ++v) {
        int k = (v < 4) ? (2 * v) : (16 + 2 * (v - 4));
        a[2 * v]     = row[k];
        a[2 * v + 1] = row[k + 1];
    }
    return a;
}

// ---------------------------------------------------------------------------
// B-fragment (32x16 bf16, KxN) from row-major fp32 weights:
// lane L supplies column N = L&15; element s holds K = 8*(L>>4) + (s<8 ? s : 8+s) (strided loads).
// fp32 -> bf16 conversion in registers (VALU co-executes with XDL WMMA).
// ---------------------------------------------------------------------------
__device__ inline v16bf load_b_f32(const float* __restrict__ p, int stride, int lane) {
    const float* q = p + (lane & 15) + (size_t)((lane >> 4) << 3) * stride;
    v16bf b;
#pragma unroll
    for (int s = 0; s < 16; ++s) {
        int k = (s < 8) ? s : (16 + (s - 8));
        b[s] = (__bf16)q[(size_t)k * stride];
    }
    return b;
}

__device__ inline v8f wmma_bf16(v16bf a, v16bf b, v8f c) {
    return __builtin_amdgcn_wmma_f32_16x16x32_bf16(
        /*neg_a=*/false, a, /*neg_b=*/false, b,
        /*c_mod=*/(short)0, c, /*reuse_a=*/false, /*reuse_b=*/false);
}

// ---------------------------------------------------------------------------
// Router: one wave per token. logits = x @ router_w + b ; top-2 softmax;
// writes dense router_scores and per-expert compacted token lists.
// ---------------------------------------------------------------------------
__global__ void __launch_bounds__(256)
router_kernel(const float* __restrict__ x, const float* __restrict__ rw,
              const float* __restrict__ rb, float* __restrict__ scores,
              int* __restrict__ counts, int* __restrict__ elist,
              float* __restrict__ ewt) {
    int wave = threadIdx.x >> 5;
    int lane = threadIdx.x & 31;
    int t = blockIdx.x * 8 + wave;

    float acc[E];
#pragma unroll
    for (int e = 0; e < E; ++e) acc[e] = 0.0f;

    for (int h = lane; h < H; h += 32) {
        float xv = x[t * H + h];
        const float* r = rw + h * E;
#pragma unroll
        for (int e = 0; e < E; ++e) acc[e] += xv * r[e];
    }
#pragma unroll
    for (int e = 0; e < E; ++e) {
#pragma unroll
        for (int off = 16; off > 0; off >>= 1)
            acc[e] += __shfl_xor(acc[e], off, 32);
    }

    if (lane == 0) {
        float v[E];
#pragma unroll
        for (int e = 0; e < E; ++e) v[e] = acc[e] + rb[e];

        int bi = 0; float bv = v[0];
#pragma unroll
        for (int e = 1; e < E; ++e) if (v[e] > bv) { bv = v[e]; bi = e; }
        int si = (bi == 0) ? 1 : 0; float sv = v[si];
#pragma unroll
        for (int e = 0; e < E; ++e)
            if (e != bi && v[e] > sv) { sv = v[e]; si = e; }

        // softmax over (bv, sv), bv >= sv
        float w0 = 1.0f / (1.0f + __expf(sv - bv));
        float w1 = 1.0f - w0;

#pragma unroll
        for (int e = 0; e < E; ++e)
            scores[t * E + e] = (e == bi) ? w0 : ((e == si) ? w1 : 0.0f);

        int p0 = atomicAdd(&counts[bi], 1);
        elist[bi * T + p0] = t; ewt[bi * T + p0] = w0;
        int p1 = atomicAdd(&counts[si], 1);
        elist[si * T + p1] = t; ewt[si * T + p1] = w1;
    }
}

// ---------------------------------------------------------------------------
// Fused expert MLP: block = (32-token tile, expert). 16 waves.
// Each weight B-fragment feeds 2 WMMAs (both M-halves); 4 per k-step with gate/up.
// GEMM1 -> clamp + GLU activation -> LDS -> GEMM2 -> weighted atomic scatter.
// ---------------------------------------------------------------------------
__global__ void __launch_bounds__(NTHREADS)
moe_kernel(const float* __restrict__ x, const float* __restrict__ guw,
           const float* __restrict__ gub, const float* __restrict__ dww,
           const float* __restrict__ dwb, float* __restrict__ out,
           const int* __restrict__ counts, const int* __restrict__ elist,
           const float* __restrict__ ewt) {
    const int e = blockIdx.y;
    const int n = counts[e];
    const int base = blockIdx.x * TILE_M;
    if (base >= n) return;
    const int valid = min(TILE_M, n - base);

    const int tid  = threadIdx.x;
    const int wave = tid >> 5;
    const int lane = tid & 31;

    extern __shared__ char smem_raw[];
    __bf16* s_x   = reinterpret_cast<__bf16*>(smem_raw);          // 32 x H  (64 KB)
    __bf16* s_act = s_x + TILE_M * H;                             // 32 x I (128 KB)
    __shared__ int   s_tok[TILE_M];
    __shared__ float s_wt[TILE_M];

    if (tid < TILE_M) {
        if (tid < valid) {
            s_tok[tid] = elist[e * T + base + tid];
            s_wt[tid]  = ewt[e * T + base + tid];
        } else {
            s_tok[tid] = 0;
            s_wt[tid]  = 0.0f;
        }
    }
    // Stage X tile (fp32 -> bf16); pad invalid rows with zero.
    for (int idx = tid; idx < TILE_M * H; idx += NTHREADS) {
        int m = idx >> 10;           // / H
        int h = idx & (H - 1);
        float v = 0.0f;
        if (m < valid) v = x[(size_t)elist[e * T + base + m] * H + h];
        s_x[m * H + h] = (__bf16)v;
    }
    __syncthreads();

    const size_t guw_e = (size_t)e * H * (2 * I);

    // ---- GEMM1: (32 x H) x (H x 2I); gate/up + both M-halves share B loads.
    for (int ct = wave; ct < I / 16; ct += NTHREADS / 32) {
        const int f0 = ct * 16;
        v8f ag0 = {}, au0 = {}, ag1 = {}, au1 = {};
        for (int k0 = 0; k0 < H; k0 += 32) {
            v16bf a0 = load_a_lds(s_x, H, lane, k0, 0);
            v16bf a1 = load_a_lds(s_x, H, lane, k0, 16);
            const float* pg = guw + guw_e + (size_t)k0 * (2 * I) + f0;
            const float* pu = pg + I;
            if (k0 + 32 < H) {
                __builtin_prefetch(pg + 32 * (2 * I), 0, 0);
                __builtin_prefetch(pu + 32 * (2 * I), 0, 0);
            }
            v16bf bg = load_b_f32(pg, 2 * I, lane);
            v16bf bu = load_b_f32(pu, 2 * I, lane);
            ag0 = wmma_bf16(a0, bg, ag0);
            ag1 = wmma_bf16(a1, bg, ag1);
            au0 = wmma_bf16(a0, bu, au0);
            au1 = wmma_bf16(a1, bu, au1);
        }
        // activation: gu=min(gu,L); up=clamp(up,-L,L); act=(up+1)*g*sigmoid(a*g)
        const int col = lane & 15;
        const float gb = gub[(size_t)e * 2 * I + f0 + col];
        const float ub = gub[(size_t)e * 2 * I + I + f0 + col];
#pragma unroll
        for (int half = 0; half < 2; ++half) {
            v8f& ag = half ? ag1 : ag0;
            v8f& au = half ? au1 : au0;
#pragma unroll
            for (int j = 0; j < 8; ++j) {
                int m = j + ((lane >> 4) << 3) + half * 16;
                float g = fminf(ag[j] + gb, LIMIT);
                float u = fmaxf(fminf(au[j] + ub, LIMIT), -LIMIT);
                float glu = g / (1.0f + __expf(-ALPHA * g));
                s_act[m * I + f0 + col] = (__bf16)((u + 1.0f) * glu);
            }
        }
    }
    __syncthreads();

    // ---- GEMM2: (32 x I) x (I x H); both M-halves share B loads.
    const size_t dww_e = (size_t)e * I * H;
    for (int ht = wave; ht < H / 16; ht += NTHREADS / 32) {
        const int h0 = ht * 16;
        v8f ac0 = {}, ac1 = {};
        for (int k0 = 0; k0 < I; k0 += 32) {
            v16bf a0 = load_a_lds(s_act, I, lane, k0, 0);
            v16bf a1 = load_a_lds(s_act, I, lane, k0, 16);
            const float* pb = dww + dww_e + (size_t)k0 * H + h0;
            if (k0 + 32 < I) __builtin_prefetch(pb + 32 * H, 0, 0);
            v16bf b = load_b_f32(pb, H, lane);
            ac0 = wmma_bf16(a0, b, ac0);
            ac1 = wmma_bf16(a1, b, ac1);
        }
        const int col = lane & 15;
        const float db = dwb[(size_t)e * H + h0 + col];
#pragma unroll
        for (int half = 0; half < 2; ++half) {
            v8f& ac = half ? ac1 : ac0;
#pragma unroll
            for (int j = 0; j < 8; ++j) {
                int m = j + ((lane >> 4) << 3) + half * 16;
                if (m < valid) {
                    int   t = s_tok[m];
                    float w = s_wt[m];
                    atomicAdd(&out[(size_t)t * H + h0 + col], w * (ac[j] + db));
                }
            }
        }
    }
}

// ---------------------------------------------------------------------------
extern "C" void kernel_launch(void* const* d_in, const int* in_sizes, int n_in,
                              void* d_out, int out_size, void* d_ws, size_t ws_size,
                              hipStream_t stream) {
    const float* x   = (const float*)d_in[0];  // (B,S,H)
    const float* rw  = (const float*)d_in[1];  // (H,E)
    const float* rb  = (const float*)d_in[2];  // (E,)
    const float* guw = (const float*)d_in[3];  // (E,H,2I)
    const float* gub = (const float*)d_in[4];  // (E,2I)
    const float* dww = (const float*)d_in[5];  // (E,I,H)
    const float* dwb = (const float*)d_in[6];  // (E,H)

    float* out    = (float*)d_out;              // T*H
    float* scores = out + (size_t)T * H;        // T*E

    int*   counts = (int*)d_ws;                                   // E ints (padded to 256B)
    int*   elist  = (int*)((char*)d_ws + 256);                    // E*T ints
    float* ewt    = (float*)((char*)d_ws + 256 + (size_t)E * T * sizeof(int));

    hipMemsetAsync(d_ws, 0, 256, stream);
    hipMemsetAsync(d_out, 0, (size_t)T * H * sizeof(float), stream);

    router_kernel<<<T / 8, 256, 0, stream>>>(x, rw, rb, scores, counts, elist, ewt);

    size_t smem = (size_t)TILE_M * H * sizeof(__bf16) +
                  (size_t)TILE_M * I * sizeof(__bf16);            // 192 KB
    moe_kernel<<<dim3(T / TILE_M, E), NTHREADS, smem, stream>>>(
        x, guw, gub, dww, dwb, out, counts, elist, ewt);
}
